// AttentionBlock_44590350467220
// MI455X (gfx1250) — matrix operand order repaired and checked by reading, not compiled, not run on hardware
//
#include <hip/hip_runtime.h>

// ---------------------------------------------------------------------------
// Types for CDNA5 WMMA (wave32): v_wmma_f32_16x16x32_bf16
// ---------------------------------------------------------------------------
typedef __attribute__((ext_vector_type(16))) __bf16 v16bf;
typedef __attribute__((ext_vector_type(8)))  float  v8f;
typedef __attribute__((ext_vector_type(4)))  unsigned int v4u;
typedef __attribute__((ext_vector_type(8)))  int  v8i;
typedef __attribute__((ext_vector_type(4)))  int  v4i;

#if defined(__has_builtin)
#if __has_builtin(__builtin_amdgcn_tensor_load_to_lds) && __has_builtin(__builtin_amdgcn_s_wait_tensorcnt)
#define HAVE_TDM 1
#endif
#endif

union Frag16 {
    unsigned short u[16];
    uint4          q[2];
    v16bf          v;
};

static __device__ inline v8f wmma_bf16(const Frag16& a, const Frag16& b, v8f c) {
    // D = A(16x32 bf16) * B(32x16 bf16) + C(16x16 f32)
    return __builtin_amdgcn_wmma_f32_16x16x32_bf16(
        /*neg_a=*/false, a.v, /*neg_b=*/false, b.v,
        /*c_mod=*/(short)0, c, /*reuse_a=*/false, /*reuse_b=*/false);
}

static __device__ inline unsigned short f2bf(float f) {
    unsigned int u = __float_as_uint(f);
    u += 0x7FFFu + ((u >> 16) & 1u);   // round-to-nearest-even
    return (unsigned short)(u >> 16);
}

static __device__ inline unsigned lds_off(const void* p) {
    // addrspace(3) pointers carry the LDS byte offset in the low 32 bits
    return (unsigned)(unsigned long long)p;
}

#ifdef HAVE_TDM
// ---------------------------------------------------------------------------
// Tensor Data Mover: 2-D tile load (bf16 elements) per D# spec (08_async_tensor
// §8.3/8.4).  tile_x elems/row, tile_y rows, stride_x = row pitch in elems.
// Tensor dims == tile dims (tiles are always fully in-bounds here).
// ---------------------------------------------------------------------------
static __device__ inline void tdm_load_2d(unsigned lds_addr, const void* gptr,
                                          unsigned tile_x, unsigned tile_y,
                                          unsigned stride_x) {
    unsigned long long ga = (unsigned long long)gptr;
    v4u g0;
    g0[0] = 1u;                                                   // count=1
    g0[1] = lds_addr;                                             // lds_addr @63:32
    g0[2] = (unsigned)ga;                                         // gaddr lo
    g0[3] = ((unsigned)(ga >> 32) & 0x01FFFFFFu) | (2u << 30);    // gaddr hi | type=2
    v8i g1;
    g1[0] = (int)(1u << 16);                                      // data_size=2B, wg_mask=0
    g1[1] = (int)((tile_x & 0xFFFFu) << 16);                      // tensor_dim0[15:0]
    g1[2] = (int)(((tile_x >> 16) & 0xFFFFu) |
                  ((tile_y & 0xFFFFu) << 16));                    // dim0 hi | dim1 lo
    g1[3] = (int)(((tile_y >> 16) & 0xFFFFu) |
                  ((tile_x & 0xFFFFu) << 16));                    // dim1 hi | tile_dim0
    g1[4] = (int)(tile_y & 0xFFFFu);                              // tile_dim1, tile_dim2=0
    g1[5] = (int)stride_x;                                        // tensor_dim0_stride lo
    g1[6] = 0;
    g1[7] = 0;
    v4i z;  z[0] = 0; z[1] = 0; z[2] = 0; z[3] = 0;
#if __clang_major__ >= 23
    v8i z8; z8[0] = 0; z8[1] = 0; z8[2] = 0; z8[3] = 0;
            z8[4] = 0; z8[5] = 0; z8[6] = 0; z8[7] = 0;
    __builtin_amdgcn_tensor_load_to_lds(g0, g1, z, z, z8, 0);
#else
    __builtin_amdgcn_tensor_load_to_lds(g0, g1, z, z, 0);
#endif
}
#endif // HAVE_TDM

// Problem constants
#define BB 4
#define SS 2048
#define DD 1024
#define HH 8
#define DK 64
#define DV 128
#define NQK 512            // H*DK

// ---------------------------------------------------------------------------
// Phase 1a: x (f32) -> xw (bf16), row-major [B*S][D]
// ---------------------------------------------------------------------------
__global__ __launch_bounds__(256) void convert_x(const float* __restrict__ x,
                                                 unsigned short* __restrict__ xw) {
    size_t i = ((size_t)blockIdx.x * 256 + threadIdx.x) * 4;
    float4 f = *(const float4*)(x + i);
    ushort4 o;
    o.x = f2bf(f.x); o.y = f2bf(f.y); o.z = f2bf(f.z); o.w = f2bf(f.w);
    *(ushort4*)(xw + i) = o;
}

// ---------------------------------------------------------------------------
// Phase 1b: fused transposed weight Wt[n][d], n in [0,2048)
// ---------------------------------------------------------------------------
__global__ __launch_bounds__(256) void build_wt(const float* __restrict__ Mq,
                                                const float* __restrict__ Mk,
                                                const float* __restrict__ Mv,
                                                unsigned short* __restrict__ wt) {
    size_t idx = (size_t)blockIdx.x * 256 + threadIdx.x;   // < 2048*1024
    int n = (int)(idx >> 10);
    int d = (int)(idx & 1023);
    float w;
    if (n < NQK)            w = Mq[(size_t)d * NQK + n];
    else if (n < 2 * NQK)   w = Mk[(size_t)d * NQK + (n - NQK)];
    else                    w = Mv[(size_t)d * (HH * DV) + (n - 2 * NQK)];
    wt[(size_t)n * DD + d] = f2bf(w);
}

// ---------------------------------------------------------------------------
// Phase 2: QKV GEMM.  C[8192x2048] = xw[8192x1024] * Wt^T
// 128x128 tile / WG, 8 waves (16x128 strip each), K-step 32.
// TDM double-buffered LDS tiles: DMA of tile k+1 overlaps WMMA on tile k.
// ---------------------------------------------------------------------------
__global__ __launch_bounds__(256) void qkv_gemm(const unsigned short* __restrict__ xw,
                                                const unsigned short* __restrict__ wt,
                                                unsigned short* __restrict__ Qb,
                                                unsigned short* __restrict__ Kb,
                                                unsigned short* __restrict__ Vt) {
    __shared__ __align__(16) unsigned short ldsA[2][128 * 32];
    __shared__ __align__(16) unsigned short ldsB[2][128 * 32];

    const int nt = blockIdx.x;           // 0..15
    const int mt = blockIdx.y;           // 0..63
    const int tid  = threadIdx.x;
    const int wave = tid >> 5;
    const int lane = tid & 31;
    const int lh = lane >> 4;
    const int ll = lane & 15;

    const unsigned short* Abase = xw + (size_t)mt * 128 * DD;
    const unsigned short* Bbase = wt + (size_t)nt * 128 * DD;

    v8f acc[8];
#pragma unroll
    for (int j = 0; j < 8; ++j)
#pragma unroll
        for (int e = 0; e < 8; ++e) acc[j][e] = 0.0f;

#ifndef HAVE_TDM
    const int ldr = tid >> 1;
    const int ldc = (tid & 1) * 16;
#endif

    // ---- prologue: fill buffer 0 ----
#ifdef HAVE_TDM
    if (wave == 0) {
        tdm_load_2d(lds_off(ldsA[0]), Abase, 32, 128, DD);
        tdm_load_2d(lds_off(ldsB[0]), Bbase, 32, 128, DD);
    }
#else
    {
        const uint4* sa = (const uint4*)(Abase + (size_t)ldr * DD + ldc);
        uint4* da = (uint4*)(ldsA[0] + ldr * 32 + ldc);
        da[0] = sa[0]; da[1] = sa[1];
        const uint4* sb = (const uint4*)(Bbase + (size_t)ldr * DD + ldc);
        uint4* db = (uint4*)(ldsB[0] + ldr * 32 + ldc);
        db[0] = sb[0]; db[1] = sb[1];
    }
#endif

    int buf = 0;
    for (int k0 = 0; k0 < DD; k0 += 32) {
#ifdef HAVE_TDM
        if (wave == 0) __builtin_amdgcn_s_wait_tensorcnt(0);
#endif
        __syncthreads();   // buffer `buf` ready; everyone done with buf^1

        if (k0 + 32 < DD) {
#ifdef HAVE_TDM
            if (wave == 0) {
                tdm_load_2d(lds_off(ldsA[buf ^ 1]), Abase + k0 + 32, 32, 128, DD);
                tdm_load_2d(lds_off(ldsB[buf ^ 1]), Bbase + k0 + 32, 32, 128, DD);
            }
#else
            const uint4* sa = (const uint4*)(Abase + (size_t)ldr * DD + k0 + 32 + ldc);
            uint4* da = (uint4*)(ldsA[buf ^ 1] + ldr * 32 + ldc);
            da[0] = sa[0]; da[1] = sa[1];
            const uint4* sb = (const uint4*)(Bbase + (size_t)ldr * DD + k0 + 32 + ldc);
            uint4* db = (uint4*)(ldsB[buf ^ 1] + ldr * 32 + ldc);
            db[0] = sb[0]; db[1] = sb[1];
#endif
        }

        // ---- compute on buffer `buf` ----
        const unsigned short* A = ldsA[buf];
        const unsigned short* Bt = ldsB[buf];
        Frag16 fa;
        {
            const unsigned short* ar = A + (wave * 16 + ll) * 32;
            fa.q[0] = *(const uint4*)(ar + lh * 8);
            fa.q[1] = *(const uint4*)(ar + 16 + lh * 8);
        }
        Frag16 fb[8];
#pragma unroll
        for (int j = 0; j < 8; ++j) {
            const unsigned short* br = Bt + (j * 16 + ll) * 32 + lh * 8;
            fb[j].q[0] = *(const uint4*)(br);
            fb[j].q[1] = *(const uint4*)(br + 16);
        }
#pragma unroll
        for (int j = 0; j < 8; ++j)
            acc[j] = wmma_bf16(fa, fb[j], acc[j]);

        buf ^= 1;
    }

    // ---- epilogue: row M = lh*8+v, col N = ll within 16x16 tile j ----
#pragma unroll
    for (int j = 0; j < 8; ++j) {
#pragma unroll
        for (int v = 0; v < 8; ++v) {
            float val = acc[j][v];
            int gm = mt * 128 + wave * 16 + lh * 8 + v;
            int gn = nt * 128 + j * 16 + ll;
            int b = gm >> 11, s = gm & 2047;
            if (gn < NQK) {
                int h = gn >> 6, k = gn & 63;
                Qb[(((size_t)(b * HH + h) * SS + s) * DK) + k] = f2bf(val * 0.125f);
            } else if (gn < 2 * NQK) {
                int g = gn - NQK;
                int h = g >> 6, k = g & 63;
                Kb[(((size_t)(b * HH + h) * SS + s) * DK) + k] = f2bf(val);
            } else {
                int g = gn - 2 * NQK;
                int h = g >> 7, dv = g & 127;
                Vt[(((size_t)(b * HH + h) * DV + dv) * SS) + s] = f2bf(val);
            }
        }
    }
}

// ---------------------------------------------------------------------------
// Phase 3: flash attention per (b,h).  Grid: x = query tile (16), y = b*H+h.
// 8 waves/WG, wave owns 16 queries.  32-key tiles, TDM double-buffered K/V^T.
// ---------------------------------------------------------------------------
__global__ __launch_bounds__(256) void attn(const unsigned short* __restrict__ Qb,
                                            const unsigned short* __restrict__ Kb,
                                            const unsigned short* __restrict__ Vt,
                                            float* __restrict__ out) {
    __shared__ __align__(16) unsigned short ldsK[2][32 * 64];     // [key][kdim]
    __shared__ __align__(16) unsigned short ldsV[2][128 * 32];    // [dv][key]
    __shared__ __align__(16) unsigned short ldsP[8][16 * 32];     // per-wave P staging

    const int qt = blockIdx.x;            // 0..15
    const int bh = blockIdx.y;            // 0..31
    const int h  = bh & 7;
    const int b  = bh >> 3;

    const unsigned short* Qh = Qb + (size_t)bh * SS * DK;
    const unsigned short* Kh = Kb + (size_t)bh * SS * DK;
    const unsigned short* Vh = Vt + (size_t)bh * DV * SS;

    const int tid  = threadIdx.x;
    const int wave = tid >> 5;
    const int lane = tid & 31;
    const int lh = lane >> 4;
    const int ll = lane & 15;
    const int q0 = qt * 128 + wave * 16;

    // Q fragments (K-dim 64 = two 16x32 A-fragments), loaded once
    Frag16 aq[2];
    {
        const unsigned short* qr = Qh + (size_t)(q0 + ll) * DK;
        aq[0].q[0] = *(const uint4*)(qr + lh * 8);
        aq[0].q[1] = *(const uint4*)(qr + 16 + lh * 8);
        aq[1].q[0] = *(const uint4*)(qr + 32 + lh * 8);
        aq[1].q[1] = *(const uint4*)(qr + 48 + lh * 8);
    }

    v8f acc[8];
#pragma unroll
    for (int j = 0; j < 8; ++j)
#pragma unroll
        for (int e = 0; e < 8; ++e) acc[j][e] = 0.0f;
    float mrow[8], lrow[8];
#pragma unroll
    for (int v = 0; v < 8; ++v) { mrow[v] = -__builtin_inff(); lrow[v] = 0.0f; }

    // ---- prologue: fill buffer 0 with first key tile ----
#ifdef HAVE_TDM
    if (wave == 0) {
        tdm_load_2d(lds_off(ldsK[0]), Kh, 64, 32, DK);
        tdm_load_2d(lds_off(ldsV[0]), Vh, 32, 128, SS);
    }
#else
    {
        int kr = tid >> 3, kc = (tid & 7) * 8;
        *(uint4*)(ldsK[0] + kr * 64 + kc) = *(const uint4*)(Kh + (size_t)kr * DK + kc);
        int vr = tid >> 1, vc = (tid & 1) * 16;
        const uint4* src = (const uint4*)(Vh + (size_t)vr * SS + vc);
        uint4* dst = (uint4*)(ldsV[0] + vr * 32 + vc);
        dst[0] = src[0]; dst[1] = src[1];
    }
#endif

    int buf = 0;
    for (int kk = 0; kk < SS; kk += 32) {
#ifdef HAVE_TDM
        if (wave == 0) __builtin_amdgcn_s_wait_tensorcnt(0);
#endif
        __syncthreads();   // buffer `buf` ready; everyone done with buf^1

        if (kk + 32 < SS) {
#ifdef HAVE_TDM
            if (wave == 0) {
                tdm_load_2d(lds_off(ldsK[buf ^ 1]), Kh + (size_t)(kk + 32) * DK, 64, 32, DK);
                tdm_load_2d(lds_off(ldsV[buf ^ 1]), Vh + (size_t)(kk + 32), 32, 128, SS);
            }
#else
            int kr = tid >> 3, kc = (tid & 7) * 8;
            *(uint4*)(ldsK[buf ^ 1] + kr * 64 + kc) =
                *(const uint4*)(Kh + (size_t)(kk + 32 + kr) * DK + kc);
            int vr = tid >> 1, vc = (tid & 1) * 16;
            const uint4* src = (const uint4*)(Vh + (size_t)vr * SS + kk + 32 + vc);
            uint4* dst = (uint4*)(ldsV[buf ^ 1] + vr * 32 + vc);
            dst[0] = src[0]; dst[1] = src[1];
#endif
        }

        const unsigned short* Kt = ldsK[buf];
        const unsigned short* Vtl = ldsV[buf];

        // ---- scores: two 16-key halves, K-dim 64 ----
        v8f s_lo, s_hi;
#pragma unroll
        for (int e = 0; e < 8; ++e) { s_lo[e] = 0.0f; s_hi[e] = 0.0f; }
        {
            Frag16 fk[4];
#pragma unroll
            for (int t = 0; t < 2; ++t) {
                const unsigned short* krow = Kt + (t * 16 + ll) * 64;
                fk[2 * t + 0].q[0] = *(const uint4*)(krow + lh * 8);
                fk[2 * t + 0].q[1] = *(const uint4*)(krow + 16 + lh * 8);
                fk[2 * t + 1].q[0] = *(const uint4*)(krow + 32 + lh * 8);
                fk[2 * t + 1].q[1] = *(const uint4*)(krow + 48 + lh * 8);
            }
            s_lo = wmma_bf16(aq[0], fk[0], s_lo);
            s_lo = wmma_bf16(aq[1], fk[1], s_lo);
            s_hi = wmma_bf16(aq[0], fk[2], s_hi);
            s_hi = wmma_bf16(aq[1], fk[3], s_hi);
        }

        // ---- online softmax over these 32 keys ----
#pragma unroll
        for (int v = 0; v < 8; ++v) {
            float mx = fmaxf(s_lo[v], s_hi[v]);
#pragma unroll
            for (int msk = 1; msk < 16; msk <<= 1)
                mx = fmaxf(mx, __shfl_xor(mx, msk, 32));
            float mnew = fmaxf(mrow[v], mx);
            float corr = __expf(mrow[v] - mnew);
            float pl = __expf(s_lo[v] - mnew);
            float ph = __expf(s_hi[v] - mnew);
            float rs = pl + ph;
#pragma unroll
            for (int msk = 1; msk < 16; msk <<= 1)
                rs += __shfl_xor(rs, msk, 32);
            lrow[v] = lrow[v] * corr + rs;
            mrow[v] = mnew;
#pragma unroll
            for (int j = 0; j < 8; ++j) acc[j][v] *= corr;
            // stage P: C-layout -> A-layout lane transpose through LDS
            ldsP[wave][(lh * 8 + v) * 32 + ll]      = f2bf(pl);
            ldsP[wave][(lh * 8 + v) * 32 + 16 + ll] = f2bf(ph);
        }
        // cross-lane LDS dependency within the wave: explicit DS wait
        asm volatile("s_wait_dscnt 0" ::: "memory");

        // ---- P (16q x 32key) x V^T (32key x 128dv) ----
        Frag16 fp;
        {
            const unsigned short* pr = &ldsP[wave][ll * 32];
            fp.q[0] = *(const uint4*)(pr + lh * 8);
            fp.q[1] = *(const uint4*)(pr + 16 + lh * 8);
        }
#pragma unroll
        for (int j = 0; j < 8; ++j) {
            Frag16 fv;
            const unsigned short* vr = Vtl + (j * 16 + ll) * 32 + lh * 8;
            fv.q[0] = *(const uint4*)(vr);
            fv.q[1] = *(const uint4*)(vr + 16);
            acc[j] = wmma_bf16(fp, fv, acc[j]);
        }

        buf ^= 1;
    }

    // ---- finalize: divide by softmax denom, write out[b][s][h*128+dv] ----
#pragma unroll
    for (int v = 0; v < 8; ++v) {
        float inv = 1.0f / lrow[v];
        int s = q0 + lh * 8 + v;
#pragma unroll
        for (int j = 0; j < 8; ++j) {
            int col = h * DV + j * 16 + ll;
            out[((size_t)b * SS + s) * (HH * DV) + col] = acc[j][v] * inv;
        }
    }
}

// ---------------------------------------------------------------------------
// Launch
// ---------------------------------------------------------------------------
extern "C" void kernel_launch(void* const* d_in, const int* in_sizes, int n_in,
                              void* d_out, int out_size, void* d_ws, size_t ws_size,
                              hipStream_t stream) {
    const float* x  = (const float*)d_in[0];
    const float* Mq = (const float*)d_in[1];
    const float* Mk = (const float*)d_in[2];
    const float* Mv = (const float*)d_in[3];
    float* out = (float*)d_out;

    unsigned short* ws = (unsigned short*)d_ws;
    unsigned short* xw = ws;                                   // 8192*1024
    unsigned short* wt = xw + (size_t)8192 * 1024;             // 2048*1024
    unsigned short* Qb = wt + (size_t)2048 * 1024;             // 4*8*2048*64
    unsigned short* Kb = Qb + (size_t)BB * HH * SS * DK;
    unsigned short* Vt = Kb + (size_t)BB * HH * SS * DK;       // 4*8*128*2048

    hipLaunchKernelGGL(convert_x, dim3(8192), dim3(256), 0, stream, x, xw);
    hipLaunchKernelGGL(build_wt,  dim3(8192), dim3(256), 0, stream, Mq, Mk, Mv, wt);
    hipLaunchKernelGGL(qkv_gemm,  dim3(16, 64), dim3(256), 0, stream, xw, wt, Qb, Kb, Vt);
    hipLaunchKernelGGL(attn,      dim3(16, 32), dim3(256), 0, stream, Qb, Kb, Vt, out);
}
